// HieNNClassifier_66417374265542
// MI455X (gfx1250) — matrix operand order repaired and checked by reading, not compiled, not run on hardware
//
#include <hip/hip_runtime.h>
#include <hip/hip_bf16.h>
#include <math.h>

#define B_    32
#define T_    4096
#define E_    128
#define H_    256
#define C_    20
#define SMAX_ 128

typedef __bf16 bf16_t;
typedef __attribute__((ext_vector_type(16))) __bf16 v16bf;
typedef __attribute__((ext_vector_type(8)))  float  v8f;

union Frag { uint4 q[2]; v16bf v; };

__device__ __forceinline__ unsigned bf1(float f) {      // f32 -> bf16 bits (RNE)
    unsigned u = __builtin_bit_cast(unsigned, f);
    u += 0x7FFFu + ((u >> 16) & 1u);
    return u >> 16;
}
__device__ __forceinline__ unsigned bfpack2(float a, float b) {
    return bf1(a) | (bf1(b) << 16);
}
__device__ __forceinline__ bf16_t f2bf(float f) {
    unsigned short h = (unsigned short)bf1(f);
    return __builtin_bit_cast(bf16_t, h);
}
__device__ __forceinline__ float bf2f(bf16_t v) {
    unsigned u = (unsigned)__builtin_bit_cast(unsigned short, v) << 16;
    return __builtin_bit_cast(float, u);
}
// branchless tanh: 1 - 2/(exp2(2*log2e*x)+1); inf/0 propagation saturates to +/-1
__device__ __forceinline__ float fast_tanh(float x) {
    float e = __builtin_amdgcn_exp2f(x * 2.8853900817779268f);
    return 1.0f - 2.0f * __builtin_amdgcn_rcpf(e + 1.0f);
}
__device__ __forceinline__ void atomAddGlobal(float* p, float v) {
    __hip_atomic_fetch_add(p, v, __ATOMIC_RELAXED, __HIP_MEMORY_SCOPE_AGENT);
}
__device__ __forceinline__ void atomAddLds(float* p, float v) {
    __hip_atomic_fetch_add(p, v, __ATOMIC_RELAXED, __HIP_MEMORY_SCOPE_WORKGROUP);
}

// ---------------------------------------------------------------------------
// Kernel 1: zero sentence accumulators; repack W1/W2 into bf16 WMMA B-fragment
// layout: pW[((kt*16+nt)*32+lane)*16 + i], k = kt*32+(lane>>4)*16+i,
// n = nt*16+(lane&15). Pair-packed dword stores.
// ---------------------------------------------------------------------------
__global__ void k_prep(const float* __restrict__ W1, const float* __restrict__ W2,
                       unsigned* __restrict__ pW1, unsigned* __restrict__ pW2,
                       float* __restrict__ sentSum) {
    int tid  = blockIdx.x * blockDim.x + threadIdx.x;
    int nthr = gridDim.x * blockDim.x;
    for (int i = tid; i < B_ * SMAX_ * H_; i += nthr) sentSum[i] = 0.f;
    for (int idx2 = tid; idx2 < (4 * 16 * 32 * 16) / 2; idx2 += nthr) {
        int idx = idx2 * 2;
        int i = idx & 15, lane = (idx >> 4) & 31, nt = (idx >> 9) & 15, kt = idx >> 13;
        int k = kt * 32 + ((lane >> 4) << 4) + i;
        int n = (nt << 4) + (lane & 15);
        pW1[idx2] = bfpack2(W1[k * H_ + n], W1[(k + 1) * H_ + n]);
    }
    for (int idx2 = tid; idx2 < (8 * 16 * 32 * 16) / 2; idx2 += nthr) {
        int idx = idx2 * 2;
        int i = idx & 15, lane = (idx >> 4) & 31, nt = (idx >> 9) & 15, kt = idx >> 13;
        int k = kt * 32 + ((lane >> 4) << 4) + i;
        int n = (nt << 4) + (lane & 15);
        pW2[idx2] = bfpack2(W2[k * H_ + n], W2[(k + 1) * H_ + n]);
    }
}

// ---------------------------------------------------------------------------
// Kernel 2: per-document scan of sentence-end markers.
// ---------------------------------------------------------------------------
__global__ __launch_bounds__(256) void k_scan(const int* __restrict__ batch_x,
                                              int* __restrict__ segW,
                                              int* __restrict__ nsentW,
                                              float* __restrict__ cntW) {
    __shared__ int sc[256];
    __shared__ int cnt_s[SMAX_];
    int b = blockIdx.x, tid = threadIdx.x;
    const int* xrow = batch_x + (size_t)b * T_;
    int t0 = tid * 16;
    int c = 0;
    for (int j = 0; j < 16; ++j) c += (xrow[t0 + j] == 1);
    sc[tid] = c; __syncthreads();
    for (int off = 1; off < 256; off <<= 1) {
        int v = sc[tid];
        int a = (tid >= off) ? sc[tid - off] : 0;
        __syncthreads();
        sc[tid] = v + a;
        __syncthreads();
    }
    int nsent = sc[255];
    int run   = sc[tid] - c;
    if (tid < SMAX_) cnt_s[tid] = 0;
    __syncthreads();
    for (int j = 0; j < 16; ++j) {
        int t = t0 + j;
        segW[(size_t)b * T_ + t] = run;
        if (run < nsent && run < SMAX_) atomicAdd(&cnt_s[run], 1);
        if (xrow[t] == 1) run++;
    }
    __syncthreads();
    if (tid < SMAX_) cntW[b * SMAX_ + tid] = (float)cnt_s[tid];
    if (tid == 0)    nsentW[b] = nsent;
}

// ---------------------------------------------------------------------------
// Kernel 3 (hot): fused gather -> bf16 WMMA GEMM -> tanh -> h tile in LDS,
// then a scalar-branch segment-pooling pass (one column per thread).
// LDS h buffer (64x264 bf16) is unioned over the A staging buffer.
// ---------------------------------------------------------------------------
#define ASTR 136   // 64 x (128+8) bf16 A staging
#define HSTR 264   // 64 x (256+8) bf16 activation tile
__global__ __launch_bounds__(256) void k_tok(const int* __restrict__ batch_x,
                                             const float* __restrict__ emb,
                                             const bf16_t* __restrict__ pW1,
                                             const float* __restrict__ b1,
                                             const int* __restrict__ segW,
                                             const int* __restrict__ nsentW,
                                             float* __restrict__ sentSum) {
    __shared__ __align__(16) bf16_t smem[64 * HSTR];        // a_s then h_s
    __shared__ int   tok_s[64];
    __shared__ int   seg_s[64];
    __shared__ float b1_s[H_];
    bf16_t* a_s = smem;
    bf16_t* h_s = smem;
    int tid = threadIdx.x;
    int b = blockIdx.x >> 6, chunk = blockIdx.x & 63;
    int tbase = chunk * 64;
    if (tid < 64) {
        int t = tbase + tid;
        tok_s[tid] = batch_x[(size_t)b * T_ + t];
        seg_s[tid] = segW[(size_t)b * T_ + t];
    }
    if (tid < H_) b1_s[tid] = b1[tid];
    __syncthreads();
    // gather + f32->bf16 stage: 64 rows x 32 float4, packed b64 LDS stores
    for (int j = 0; j < 8; ++j) {
        int fi = tid + j * 256;
        int row = fi >> 5, c4 = fi & 31;
        float4 v = *((const float4*)(emb + (size_t)tok_s[row] * E_) + c4);
        uint2 p; p.x = bfpack2(v.x, v.y); p.y = bfpack2(v.z, v.w);
        *(uint2*)(a_s + row * ASTR + c4 * 4) = p;
    }
    __syncthreads();

    int lane = tid & 31, wave = tid >> 5;
    int mt = wave & 3, ntg = wave >> 2;
    int mh = (lane >> 4) << 3;

    // A fragments for this M-tile -> registers, then a_s LDS is reusable as h_s
    Frag A[4];
    {
        int m = (mt << 4) + (lane & 15);
        for (int kt = 0; kt < 4; ++kt) {
            const bf16_t* ap = a_s + m * ASTR + kt * 32 + (mh << 1);
            A[kt].q[0] = *(const uint4*)(ap);
            A[kt].q[1] = *(const uint4*)(ap + 8);
        }
    }
    __syncthreads();                                        // all A reads done

    int n0 = lane & 15;
    int mrow = (mt << 4) + mh;                              // first of 8 rows
    for (int nt8 = 0; nt8 < 8; ++nt8) {
        int nt = ntg * 8 + nt8;
        if (nt8 + 1 < 8)
            __builtin_prefetch(pW1 + ((size_t)((nt + 1) * 32 + lane) << 4), 0, 3);
        v8f acc = {};
        for (int kt = 0; kt < 4; ++kt) {
            Frag Bf;
            const uint4* bp = (const uint4*)(pW1 + ((size_t)((kt * 16 + nt) * 32 + lane) << 4));
            Bf.q[0] = bp[0]; Bf.q[1] = bp[1];
            acc = __builtin_amdgcn_wmma_f32_16x16x32_bf16(false, A[kt].v, false, Bf.v,
                                                          (short)0, acc, false, false);
        }
        int n = (nt << 4) + n0;
        float bias = b1_s[n];
        for (int r = 0; r < 8; ++r)                          // branchless store
            h_s[(mrow + r) * HSTR + n] = f2bf(fast_tanh(acc[r] + bias));
    }
    __syncthreads();

    // segment pooling: thread tid owns column n; token walk is identical for
    // all threads -> readfirstlane makes the boundary test a scalar branch.
    int nsent = __builtin_amdgcn_readfirstlane(nsentW[b]);
    int n = tid;
    float runv = 0.f;
    int prev = __builtin_amdgcn_readfirstlane(seg_s[0]);
    for (int t = 0; t < 64; ++t) {
        int seg = __builtin_amdgcn_readfirstlane(seg_s[t]);
        if (seg != prev) {
            if (prev < nsent)
                atomAddGlobal(&sentSum[((size_t)b * SMAX_ + prev) * H_ + n], runv);
            runv = 0.f; prev = seg;
        }
        runv += bf2f(h_s[t * HSTR + n]);
    }
    if (prev < nsent)
        atomAddGlobal(&sentSum[((size_t)b * SMAX_ + prev) * H_ + n], runv);
}

// ---------------------------------------------------------------------------
// Kernel 4: sentence stage: mean scale -> bf16 WMMA GEMM (W2) -> tanh ->
// masked mean over sentences (branchless select + one ds_add per tile).
// ---------------------------------------------------------------------------
#define DSTR 264
__global__ __launch_bounds__(256) void k_sent(const float* __restrict__ sentSum,
                                              const float* __restrict__ cntW,
                                              const int* __restrict__ nsentW,
                                              const bf16_t* __restrict__ pW2,
                                              const float* __restrict__ b2,
                                              float* __restrict__ docW) {
    __shared__ __align__(16) bf16_t a_s[64 * DSTR];
    __shared__ float rc_s[SMAX_];
    __shared__ float b2_s[H_];
    __shared__ float doc_acc[H_];
    int b = blockIdx.x, tid = threadIdx.x;
    if (tid < SMAX_) { float c = cntW[b * SMAX_ + tid]; rc_s[tid] = 1.f / fmaxf(c, 1.f); }
    if (tid < H_)    { b2_s[tid] = b2[tid]; doc_acc[tid] = 0.f; }
    __syncthreads();
    int nsent = nsentW[b];
    int lane = tid & 31, wave = tid >> 5;
    int mtl = wave & 3, ntg = wave >> 2;
    int mh = (lane >> 4) << 3;

    for (int half = 0; half < 2; ++half) {
        for (int j = 0; j < 16; ++j) {                      // 64 rows x 64 float4
            int fi = tid + j * 256;
            int row = fi >> 6, c4 = fi & 63;
            int grow = half * 64 + row;
            float4 v = *((const float4*)(sentSum + ((size_t)b * SMAX_ + grow) * H_) + c4);
            float rc = rc_s[grow];
            uint2 p; p.x = bfpack2(v.x * rc, v.y * rc); p.y = bfpack2(v.z * rc, v.w * rc);
            *(uint2*)(a_s + row * DSTR + c4 * 4) = p;
        }
        __syncthreads();
        int m = (mtl << 4) + (lane & 15);
        Frag A8[8];
        for (int kt = 0; kt < 8; ++kt) {
            const bf16_t* ap = a_s + m * DSTR + kt * 32 + (mh << 1);
            A8[kt].q[0] = *(const uint4*)ap;
            A8[kt].q[1] = *(const uint4*)(ap + 8);
        }
        for (int nt8 = 0; nt8 < 8; ++nt8) {
            int nt = ntg * 8 + nt8;
            v8f acc = {};
            for (int kt = 0; kt < 8; ++kt) {
                Frag Bf;
                const uint4* bp = (const uint4*)(pW2 + ((size_t)((kt * 16 + nt) * 32 + lane) << 4));
                Bf.q[0] = bp[0]; Bf.q[1] = bp[1];
                acc = __builtin_amdgcn_wmma_f32_16x16x32_bf16(false, A8[kt].v, false, Bf.v,
                                                              (short)0, acc, false, false);
            }
            int n = (nt << 4) + (lane & 15);
            float bias = b2_s[n];
            float s = 0.f;                                   // all rows share n
            for (int r = 0; r < 8; ++r) {
                int mg = half * 64 + (mtl << 4) + mh + r;
                float tv = fast_tanh(acc[r] + bias);
                s += (mg < nsent) ? tv : 0.f;                // cndmask, no branch
            }
            atomAddLds(&doc_acc[n], s);
        }
        __syncthreads();
    }
    if (tid < H_) docW[b * H_ + tid] = doc_acc[tid] / fmaxf((float)nsent, 1.f);
}

// ---------------------------------------------------------------------------
// Kernel 5: classifier head + log_softmax (tiny).
// ---------------------------------------------------------------------------
__global__ __launch_bounds__(256) void k_head(const float* __restrict__ docW,
                                              const float* __restrict__ Wc,
                                              const float* __restrict__ bc,
                                              float* __restrict__ out) {
    __shared__ float lg[B_][C_];
    int tid = threadIdx.x;
    int b = tid >> 3, j = tid & 7;
    for (int c = j; c < C_; c += 8) {
        float acc = bc[c];
        const float* dr = docW + b * H_;
        for (int k = 0; k < H_; ++k) acc += dr[k] * Wc[k * C_ + c];
        lg[b][c] = acc;
    }
    __syncthreads();
    if (tid < B_) {
        float m = -1e30f;
        for (int c = 0; c < C_; ++c) m = fmaxf(m, lg[tid][c]);
        float s = 0.f;
        for (int c = 0; c < C_; ++c) s += expf(lg[tid][c] - m);
        float ls = logf(s);
        for (int c = 0; c < C_; ++c) out[tid * C_ + c] = lg[tid][c] - m - ls;
    }
}

extern "C" void kernel_launch(void* const* d_in, const int* in_sizes, int n_in,
                              void* d_out, int out_size, void* d_ws, size_t ws_size,
                              hipStream_t stream) {
    (void)in_sizes; (void)n_in; (void)out_size; (void)ws_size;
    const int*   batch_x = (const int*)d_in[0];
    const float* emb     = (const float*)d_in[2];
    const float* W1      = (const float*)d_in[3];
    const float* b1      = (const float*)d_in[4];
    const float* W2      = (const float*)d_in[5];
    const float* b2      = (const float*)d_in[6];
    const float* Wc      = (const float*)d_in[7];
    const float* bc      = (const float*)d_in[8];
    float* out = (float*)d_out;

    char* ws = (char*)d_ws;
    size_t off = 0;
    auto alloc = [&](size_t bytes) { char* p = ws + off; off = (off + bytes + 255) & ~(size_t)255; return p; };
    int*    segW    = (int*)   alloc((size_t)B_ * T_ * 4);
    int*    nsentW  = (int*)   alloc(B_ * 4);
    float*  cntW    = (float*) alloc((size_t)B_ * SMAX_ * 4);
    float*  sentSum = (float*) alloc((size_t)B_ * SMAX_ * H_ * 4);
    bf16_t* pW1     = (bf16_t*)alloc((size_t)E_ * H_ * 2);
    bf16_t* pW2     = (bf16_t*)alloc((size_t)H_ * H_ * 2);
    float*  docW    = (float*) alloc((size_t)B_ * H_ * 4);

    k_prep<<<512, 256, 0, stream>>>(W1, W2, (unsigned*)pW1, (unsigned*)pW2, sentSum);
    k_scan<<<B_, 256, 0, stream>>>(batch_x, segW, nsentW, cntW);
    k_tok<<<B_ * (T_ / 64), 256, 0, stream>>>(batch_x, emb, pW1, b1, segW, nsentW, sentSum);
    k_sent<<<B_, 256, 0, stream>>>(sentSum, cntW, nsentW, pW2, b2, docW);
    k_head<<<1, 256, 0, stream>>>(docW, Wc, bc, out);
}
